// RingAttention_64278480552169
// MI455X (gfx1250) — compile-verified, matrix operand order checked by reading, not run on hardware
//
#include <hip/hip_runtime.h>

typedef __bf16 bf16;
typedef __attribute__((ext_vector_type(16))) bf16     v16bf;
typedef __attribute__((ext_vector_type(8)))  bf16     v8bf;
typedef __attribute__((ext_vector_type(8)))  float    v8f;
typedef __attribute__((ext_vector_type(4)))  unsigned u32x4;
typedef __attribute__((ext_vector_type(8)))  unsigned u32x8;

#define WMMA_BF16(a, b, c) \
  __builtin_amdgcn_wmma_f32_16x16x32_bf16(false, (a), false, (b), (short)0, (c), false, false)

// LDS byte offset of a __shared__ object: low 32 bits of the flat address are
// the LDS offset (ISA 10.2: LDS aperture lives in the high 32 bits).
#define LDS_OFF(p) ((unsigned)(unsigned long long)(p))

// ds_swizzle_b32 group-of-32 XOR exchange: offset[15]=0, and_mask=0x1f,
// or_mask=0, xor_mask=XMASK (ISA §11: SWAPX1/2/4/8 patterns). One instruction
// per butterfly step, immediate-encoded (vs generic shuffle's cmp+cndmask+bpermute).
template <int XMASK>
__device__ __forceinline__ float swz_max(float v) {
  int o = __builtin_amdgcn_ds_swizzle(__float_as_int(v), (XMASK << 10) | 0x1f);
  return fmaxf(v, __int_as_float(o));
}
template <int XMASK>
__device__ __forceinline__ float swz_add(float v) {
  int o = __builtin_amdgcn_ds_swizzle(__float_as_int(v), (XMASK << 10) | 0x1f);
  return v + __int_as_float(o);
}

// ---------------------------------------------------------------------------
// TDM: issue a 2D tensor_load_to_lds (one wave should execute this).
// tile_d0 elems (contiguous, bf16), tile_d1 rows; LDS gets pad_amount DWORDs of
// padding every pad_interval-coded DWORDs (interval code: 0=2DW,1=4,2=8,3=16,
// 4=32,5=64; amount code: n => n+1 DW).
// ---------------------------------------------------------------------------
__device__ __forceinline__ void tdm_load_2d_bf16(unsigned lds_byte_addr,
                                                 const void* gaddr,
                                                 unsigned tile_d0, unsigned tile_d1,
                                                 unsigned tensor_d0, unsigned tensor_d1,
                                                 unsigned stride_elems,
                                                 unsigned pad_interval_code,
                                                 unsigned pad_amount_code) {
  unsigned long long ga = (unsigned long long)gaddr;
  u32x4 g0;
  g0[0] = 1u;                                            // count=1, user D#
  g0[1] = lds_byte_addr;                                 // lds_addr
  g0[2] = (unsigned)(ga & 0xFFFFFFFFu);                  // global_addr[31:0]
  g0[3] = (unsigned)((ga >> 32) & 0x01FFFFFFu)           // global_addr[56:32]
          | (2u << 30);                                  // type=2 ("image")
  u32x8 g1;
  g1[0] = (1u << 16)                                     // data_size = 2 bytes
          | (1u << 20)                                   // pad_enable
          | (pad_interval_code << 22)
          | (pad_amount_code << 25);
  g1[1] = (tensor_d0 & 0xFFFFu) << 16;                   // tensor_dim0[15:0] @bit48
  g1[2] = (tensor_d0 >> 16) | ((tensor_d1 & 0xFFFFu) << 16);
  g1[3] = (tensor_d1 >> 16) | ((tile_d0 & 0xFFFFu) << 16);
  g1[4] = tile_d1 & 0xFFFFu;                             // tile_dim1; tile_dim2=0
  g1[5] = stride_elems;                                  // tensor_dim0_stride[31:0]
  g1[6] = 0u;                                            // stride hi / dim1_stride
  g1[7] = 0u;
  asm volatile("tensor_load_to_lds %0, %1" :: "s"(g0), "s"(g1) : "memory");
}

__device__ __forceinline__ void tdm_wait() {
  asm volatile("s_wait_tensorcnt 0x0" ::: "memory");
}

// Async global->LDS 16B copy (GVS mode: SGPR base + per-lane 32-bit byte offset)
__device__ __forceinline__ void async_ld_b128(unsigned lds_byte_addr,
                                              unsigned gbyte_off, const void* sbase) {
  asm volatile("global_load_async_to_lds_b128 %0, %1, %2"
               :: "v"(lds_byte_addr), "v"(gbyte_off), "s"(sbase) : "memory");
}
__device__ __forceinline__ void async_wait() {
  asm volatile("s_wait_asynccnt 0x0" ::: "memory");
}

// 16x32 bf16 A-style fragment (B for the A@B^T form is identical) from a
// row-major LDS tile. ISA 7.12.2: lanes 0-15: row=lane, K={0..7,16..23};
// lanes 16-31: row=lane-16, K={8..15,24..31}. Two 16B ds loads per lane.
__device__ __forceinline__ v16bf load_frag(const bf16* base, int ld, int kbase) {
  const int lane = threadIdx.x & 31;
  const int r  = lane & 15;
  const int kb = kbase + ((lane >> 4) << 3);
  union { v16bf v; v8bf h[2]; } u;
  u.h[0] = *(const v8bf*)(base + r * ld + kb);
  u.h[1] = *(const v8bf*)(base + r * ld + kb + 16);
  return u.v;
}

// ---------------------------------------------------------------------------
// C[M,N] = A[M,K] @ W[N,K]^T  (torch Linear). bf16 in, f32 accum.
// 128x128 block tile, 8 waves, 64x32 per wave, K-step 32. Tiles arrive by TDM.
// ---------------------------------------------------------------------------
template <bool F32OUT>
__global__ __launch_bounds__(256) void gemm_xwt(const bf16* __restrict__ A,
                                                const bf16* __restrict__ W,
                                                void* __restrict__ Cout,
                                                int M, int N, int K) {
  __shared__ bf16 sA[128][40];  // 64B row + 16B TDM pad = 80B stride
  __shared__ bf16 sB[128][40];
  const int t  = threadIdx.x;
  const int m0 = blockIdx.y * 128;
  const int n0 = blockIdx.x * 128;
  const int w  = t >> 5;
  const int wm = (w >> 2) * 64;
  const int wn = (w & 3) * 32;
  const unsigned ldsA = LDS_OFF(&sA[0][0]);
  const unsigned ldsB = LDS_OFF(&sB[0][0]);

  v8f acc[4][2] = {};

  for (int k0 = 0; k0 < K; k0 += 32) {
    if (w == 0) {  // one wave drives the TDM; EXEC is ignored by tensor ops
      tdm_load_2d_bf16(ldsA, A + (size_t)m0 * K + k0, 32, 128,
                       (unsigned)K, (unsigned)M, (unsigned)K, 3, 3);
      tdm_load_2d_bf16(ldsB, W + (size_t)n0 * K + k0, 32, 128,
                       (unsigned)K, (unsigned)N, (unsigned)K, 3, 3);
      tdm_wait();
    }
    __syncthreads();

    v16bf af[4], bfr[2];
#pragma unroll
    for (int mi = 0; mi < 4; ++mi) af[mi] = load_frag(&sA[wm + mi * 16][0], 40, 0);
#pragma unroll
    for (int ni = 0; ni < 2; ++ni) bfr[ni] = load_frag(&sB[wn + ni * 16][0], 40, 0);
#pragma unroll
    for (int mi = 0; mi < 4; ++mi)
#pragma unroll
      for (int ni = 0; ni < 2; ++ni)
        acc[mi][ni] = WMMA_BF16(af[mi], bfr[ni], acc[mi][ni]);
    __syncthreads();
  }

  const int lane = t & 31;
  const int half = lane >> 4;
  const int cn   = lane & 15;
#pragma unroll
  for (int mi = 0; mi < 4; ++mi)
    for (int ni = 0; ni < 2; ++ni)
#pragma unroll
      for (int r = 0; r < 8; ++r) {
        int gm = m0 + wm + mi * 16 + r + half * 8;
        int gn = n0 + wn + ni * 16 + cn;
        float v = acc[mi][ni][r];
        if (F32OUT) ((float*)Cout)[(size_t)gm * N + gn] = v;
        else        ((bf16*)Cout)[(size_t)gm * N + gn] = (bf16)v;
      }
}

// ---------------------------------------------------------------------------
// Flash attention over per-head slices of interleaved [B,S,D] q/k/v (HD=128).
// One block = one (b,h,128-q-row tile); 8 waves, 16 q rows each; KV step 32.
// Q tile via TDM (fragments hoisted out of the KV loop); K tile via async
// global->LDS; V transposed manually.
// ---------------------------------------------------------------------------
__global__ __launch_bounds__(256) void flash_attn(const bf16* __restrict__ Q,
                                                  const bf16* __restrict__ Kk,
                                                  const bf16* __restrict__ V,
                                                  bf16* __restrict__ O,
                                                  int B_, int S_, int D_, int H_) {
  __shared__ bf16 sQ[128][136];    // 256B row + 16B TDM pad = 272B stride
  __shared__ bf16 sK[32][136];
  __shared__ bf16 sVt[128][40];    // V transposed: [d][kv]
  __shared__ bf16 sP[8][16][40];   // per-wave P tile (C->A relayout)

  const int t = threadIdx.x;
  const int lane = t & 31;
  const int w    = t >> 5;
  const int half = lane >> 4;
  const int cn   = lane & 15;
  const int qt = blockIdx.x;
  const int bh = blockIdx.y;
  const int b  = bh / H_;
  const int h  = bh % H_;
  const size_t hoff = (size_t)h * 128;
  const unsigned ldsK = LDS_OFF(&sK[0][0]);

  // Q tile [128][128] via TDM (pad 64DW interval, 4DW amount -> 272B stride)
  if (w == 0) {
    tdm_load_2d_bf16(LDS_OFF(&sQ[0][0]),
                     Q + (size_t)(b * S_ + qt * 128) * D_ + hoff,
                     128, 128, (unsigned)D_, (unsigned)(B_ * S_),
                     (unsigned)D_, 5, 3);
    tdm_wait();
  }
  __syncthreads();

  // Q A-fragments are invariant over the KV loop: hoist (4 x v16bf = 32 VGPRs)
  v16bf aq[4];
#pragma unroll
  for (int kk = 0; kk < 4; ++kk) aq[kk] = load_frag(&sQ[w * 16][0], 136, kk * 32);

  const float scale = 0.08838834764831845f;  // 1/sqrt(128)
  float mrow[8], lrow[8];
#pragma unroll
  for (int r = 0; r < 8; ++r) { mrow[r] = -1e30f; lrow[r] = 0.f; }
  v8f oacc[8] = {};

  for (int kv0 = 0; kv0 < S_; kv0 += 32) {
    // K tile [32][128] via async global->LDS, 2 x 16B per thread
#pragma unroll
    for (int i = 0; i < 2; ++i) {
      int c = t + i * 256;
      int row = c >> 4;
      int col = (c & 15) * 8;
      unsigned lds = ldsK + (unsigned)(row * 136 + col) * 2u;
      unsigned gof = (unsigned)(((size_t)(b * S_ + kv0 + row) * D_ + hoff + col) * 2u);
      async_ld_b128(lds, gof, Kk);
    }
    // V tile loaded transposed into sVt[d][kv]
    {
      int kr  = t >> 3;
      int dc0 = (t & 7) * 16;
      const bf16* vp = V + (size_t)(b * S_ + kv0 + kr) * D_ + hoff + dc0;
#pragma unroll
      for (int j = 0; j < 16; ++j) sVt[dc0 + j][kr] = vp[j];
    }
    async_wait();
    __syncthreads();

    // scores: 16 q rows x 32 kv, contract over d=128 (4 WMMA K-steps)
    v8f sf[2] = {};
#pragma unroll
    for (int ni = 0; ni < 2; ++ni)
#pragma unroll
      for (int kk = 0; kk < 4; ++kk) {
        v16bf bk = load_frag(&sK[ni * 16][0], 136, kk * 32);
        sf[ni] = WMMA_BF16(aq[kk], bk, sf[ni]);
      }
    sf[0] = sf[0] * scale;
    sf[1] = sf[1] * scale;

    // online softmax per row (row = r + half*8; reduce across 16 N-lanes via
    // ds_swizzle XOR butterflies)
#pragma unroll
    for (int r = 0; r < 8; ++r) {
      float mx = fmaxf(sf[0][r], sf[1][r]);
      mx = swz_max<1>(mx); mx = swz_max<2>(mx); mx = swz_max<4>(mx); mx = swz_max<8>(mx);
      float mnew  = fmaxf(mrow[r], mx);
      float alpha = __expf(mrow[r] - mnew);
      float p0 = __expf(sf[0][r] - mnew);
      float p1 = __expf(sf[1][r] - mnew);
      float rs = p0 + p1;
      rs = swz_add<1>(rs); rs = swz_add<2>(rs); rs = swz_add<4>(rs); rs = swz_add<8>(rs);
      lrow[r] = lrow[r] * alpha + rs;
      mrow[r] = mnew;
#pragma unroll
      for (int nt = 0; nt < 8; ++nt) oacc[nt][r] *= alpha;
      sP[w][r + half * 8][cn]      = (bf16)p0;
      sP[w][r + half * 8][16 + cn] = (bf16)p1;
    }
    // sP is wave-private: only our own DS ops need to land
    asm volatile("s_wait_dscnt 0x0" ::: "memory");

    // out += P[16x32] @ V[32x128]
    v16bf ap = load_frag(&sP[w][0][0], 40, 0);
#pragma unroll
    for (int nt = 0; nt < 8; ++nt) {
      v16bf bv = load_frag(&sVt[nt * 16][0], 40, 0);
      oacc[nt] = WMMA_BF16(ap, bv, oacc[nt]);
    }
    __syncthreads();
  }

#pragma unroll
  for (int r = 0; r < 8; ++r) {
    float inv = 1.0f / lrow[r];
    int qrow = qt * 128 + w * 16 + r + half * 8;
#pragma unroll
    for (int nt = 0; nt < 8; ++nt)
      O[(size_t)(b * S_ + qrow) * D_ + hoff + nt * 16 + cn] = (bf16)(oacc[nt][r] * inv);
  }
}

__global__ void cvt_f32_bf16(const float* __restrict__ in, bf16* __restrict__ out, int n) {
  int i = blockIdx.x * 256 + threadIdx.x;
  if (i < n) out[i] = (bf16)in[i];
}

extern "C" void kernel_launch(void* const* d_in, const int* in_sizes, int n_in,
                              void* d_out, int out_size, void* d_ws, size_t ws_size,
                              hipStream_t stream) {
  (void)in_sizes; (void)n_in; (void)out_size; (void)ws_size;
  constexpr int Bd = 2, Sd = 2048, Dd = 2048, Hd = 16;
  constexpr int NX = Bd * Sd * Dd;  // 8388608
  constexpr int NW = Dd * Dd;       // 4194304

  const float* x  = (const float*)d_in[0];
  const float* Wq = (const float*)d_in[1];
  const float* Wk = (const float*)d_in[2];
  const float* Wv = (const float*)d_in[3];
  const float* Wo = (const float*)d_in[4];

  bf16* p   = (bf16*)d_ws;
  bf16* xb  = p; p += NX;
  bf16* wqb = p; p += NW;
  bf16* wkb = p; p += NW;
  bf16* wvb = p; p += NW;
  bf16* wob = p; p += NW;
  bf16* qb  = p; p += NX;
  bf16* kb  = p; p += NX;
  bf16* vb  = p; p += NX;
  bf16* cb  = p; p += NX;

  cvt_f32_bf16<<<NX / 256, 256, 0, stream>>>(x,  xb,  NX);
  cvt_f32_bf16<<<NW / 256, 256, 0, stream>>>(Wq, wqb, NW);
  cvt_f32_bf16<<<NW / 256, 256, 0, stream>>>(Wk, wkb, NW);
  cvt_f32_bf16<<<NW / 256, 256, 0, stream>>>(Wv, wvb, NW);
  cvt_f32_bf16<<<NW / 256, 256, 0, stream>>>(Wo, wob, NW);

  dim3 gg(Dd / 128, (Bd * Sd) / 128);  // (16, 32)
  gemm_xwt<false><<<gg, 256, 0, stream>>>(xb, wqb, qb, Bd * Sd, Dd, Dd);
  gemm_xwt<false><<<gg, 256, 0, stream>>>(xb, wkb, kb, Bd * Sd, Dd, Dd);
  gemm_xwt<false><<<gg, 256, 0, stream>>>(xb, wvb, vb, Bd * Sd, Dd, Dd);

  dim3 ga(Sd / 128, Bd * Hd);  // (16, 32)
  flash_attn<<<ga, 256, 0, stream>>>(qb, kb, vb, cb, Bd, Sd, Dd, Hd);

  gemm_xwt<true><<<gg, 256, 0, stream>>>(cb, wob, d_out, Bd * Sd, Dd, Dd);
}
